// CharRNN_28321014349882
// MI455X (gfx1250) — matrix operand order repaired and checked by reading, not compiled, not run on hardware
//
#include <hip/hip_runtime.h>
#include <hip/hip_bf16.h>
#include <math.h>

// ---------- problem constants ----------
#define BB 64
#define SS 256
#define HH 1024
#define VV 8000
#define LL 2
#define K2H 2048   // 2*H
#define N4H 4096   // 4*H
#define NROWS (BB*SS)   // 16384

typedef __bf16 bf16;
typedef __attribute__((ext_vector_type(16))) __bf16 v16bf;
typedef __attribute__((ext_vector_type(8)))  __bf16 v8bf;
typedef __attribute__((ext_vector_type(8)))  float  v8f;

static __device__ __forceinline__ v8f wmma_bf16(v16bf a, v16bf b, v8f c) {
  return __builtin_amdgcn_wmma_f32_16x16x32_bf16(false, a, false, b, (short)0, c, false, false);
}
static __device__ __forceinline__ float sigf(float x) { return 1.0f / (1.0f + expf(-x)); }

// =====================================================================
// One-time prep kernels
// =====================================================================

// x[b*S+t][h] = bf16(embedding[inputs[b*S+t]][h])
__global__ __launch_bounds__(256) void k_embed(const int* __restrict__ inp,
                                               const float* __restrict__ emb,
                                               bf16* __restrict__ x) {
  int idx = blockIdx.x * 256 + threadIdx.x;        // over B*S*H = 16,777,216
  int bs = idx >> 10;
  int h  = idx & 1023;
  int tok = inp[bs];
  x[idx] = (bf16)emb[(size_t)tok * HH + h];
}

// Wt[l][n][k] = bf16(W[l][k][n])  (transpose so B-fragments are contiguous in k)
__global__ __launch_bounds__(256) void k_cast_lstmw(const float* __restrict__ w,
                                                    bf16* __restrict__ wt) {
  int idx = blockIdx.x * 256 + threadIdx.x;        // over L*2H*4H = 16,777,216
  int l   = idx / (K2H * N4H);
  int rem = idx - l * (K2H * N4H);
  int k   = rem >> 12;          // /4096
  int n   = rem & (N4H - 1);
  wt[(size_t)l * N4H * K2H + (size_t)n * K2H + k] = (bf16)w[idx];
}

// SWt[n][k] = bf16(SW[k][n])
__global__ __launch_bounds__(256) void k_cast_sw(const float* __restrict__ sw,
                                                 bf16* __restrict__ swt) {
  int idx = blockIdx.x * 256 + threadIdx.x;        // over H*V = 8,192,000
  if (idx >= HH * VV) return;
  int k = idx / VV;
  int n = idx - k * VV;
  swt[(size_t)n * HH + k] = (bf16)sw[idx];
}

__global__ __launch_bounds__(256) void k_zero_u32(unsigned int* __restrict__ p, int n) {
  int i = blockIdx.x * 256 + threadIdx.x;
  if (i < n) p[i] = 0u;
}

// =====================================================================
// Fully fused LSTM step for one layer:
//   z = concat(A0[64][1024], A1[64][1024]) @ W[2048][4096] + bias
//   i,j,f,o = split(z,4);  c' = c*sig(f+1)+sig(i)*tanh(j);  h' = sig(o)*tanh(c')
// Block owns a 16-wide h-slice and ALL 4 gate columns for it.
// 8 waves = 4 m-tiles x 2 gate-pairs; each wave: 2 accumulators, full K=2048.
// No LDS in the K loop (A is L2-resident, loaded directly); one barrier total.
// h is double-buffered by the caller (reads A1=h_prev, writes h_new).
// =====================================================================
__global__ __launch_bounds__(256) void k_lstm_step(
    const bf16* __restrict__ A0, int lda0,       // layer input rows (k<1024)
    const bf16* __restrict__ A1,                 // h_prev [64][1024]   (k>=1024)
    const bf16* __restrict__ Wt,                 // [4096][2048] transposed bf16
    const float* __restrict__ bias,              // [4096]
    float* __restrict__ c,                       // [64][1024] in-place
    bf16* __restrict__ hnew,                     // [64][1024]
    bf16* __restrict__ out2, int out2_stride) {  // optional extra h destination
  __shared__ float zbuf[4][64][16];

  const int tid  = threadIdx.x;
  const int wid  = tid >> 5;
  const int lane = tid & 31;
  const int hlf  = lane >> 4;
  const int l16  = lane & 15;
  const int mt   = wid & 3;          // m tile (16 rows)
  const int gp   = wid >> 2;         // gate pair: 0 -> gates {0,1}, 1 -> {2,3}
  const int hcol = blockIdx.x * 16 + l16;

  const bf16* bp0 = Wt + ((size_t)(2 * gp)     * HH + hcol) * K2H;
  const bf16* bp1 = Wt + ((size_t)(2 * gp + 1) * HH + hcol) * K2H;

  const int m = mt * 16 + l16;
  const bf16* arow0 = A0 + (size_t)m * lda0 + hlf * 16;
  const bf16* arow1 = A1 + (size_t)m * HH   + hlf * 16;

  v8f acc0 = {}, acc1 = {};

  // ---- k in [0,1024): input rows come from A0 ----
#pragma unroll 4
  for (int k0 = 0; k0 < HH; k0 += 32) {
    union { v16bf v; v8bf h[2]; } ua;
    ua.h[0] = *(const v8bf*)(arow0 + k0);        // K = k0 + hlf*16 .. +7
    ua.h[1] = *(const v8bf*)(arow0 + k0 + 8);    // K = k0 + 8 + hlf*16 .. +7
    v16bf b0 = *(const v16bf*)(bp0 + k0 + hlf * 16);
    v16bf b1 = *(const v16bf*)(bp1 + k0 + hlf * 16);
    acc0 = wmma_bf16(ua.v, b0, acc0);
    acc1 = wmma_bf16(ua.v, b1, acc1);
  }
  // ---- k in [1024,2048): input rows come from A1 (h_prev) ----
#pragma unroll 4
  for (int k0 = 0; k0 < HH; k0 += 32) {
    union { v16bf v; v8bf h[2]; } ua;
    ua.h[0] = *(const v8bf*)(arow1 + k0);
    ua.h[1] = *(const v8bf*)(arow1 + k0 + 8);
    v16bf b0 = *(const v16bf*)(bp0 + HH + k0 + hlf * 16);
    v16bf b1 = *(const v16bf*)(bp1 + HH + k0 + hlf * 16);
    acc0 = wmma_bf16(ua.v, b0, acc0);
    acc1 = wmma_bf16(ua.v, b1, acc1);
  }

  // ---- dump gate preactivations to LDS ----
#pragma unroll
  for (int r = 0; r < 8; ++r) {
    int row = mt * 16 + r + 8 * hlf;
    zbuf[2 * gp][row][l16]     = acc0[r];
    zbuf[2 * gp + 1][row][l16] = acc1[r];
  }
  __syncthreads();

  // ---- gate math + state update (64 rows x 16 cols = 1024 elems) ----
  for (int e = tid; e < 64 * 16; e += 256) {
    int row = e >> 4;
    int col = e & 15;
    int h   = blockIdx.x * 16 + col;
    float zi = zbuf[0][row][col] + bias[h];
    float zj = zbuf[1][row][col] + bias[HH + h];
    float zf = zbuf[2][row][col] + bias[2 * HH + h];
    float zo = zbuf[3][row][col] + bias[3 * HH + h];
    int ci = row * HH + h;
    float cv = c[ci];
    float nc = cv * sigf(zf + 1.0f) + sigf(zi) * tanhf(zj);
    float hv = sigf(zo) * tanhf(nc);
    c[ci] = nc;
    bf16 hb = (bf16)hv;
    hnew[ci] = hb;
    if (out2) out2[(size_t)row * out2_stride + h] = hb;
  }
}

// =====================================================================
// Fused logits GEMM + online log-softmax + NLL.
// Block = 128 threads (4 waves) owns 16 rows; A [16][1024] staged in 32KB LDS.
// Each wave sweeps GROUPS OF FOUR 16-col N-tiles (4 WMMAs per A-fragment
// load), keeping per-row running (max, sumexp) in registers.
// =====================================================================
__global__ __launch_bounds__(128) void k_logits_nll(const bf16* __restrict__ outs,  // [16384][1024]
                                                    const bf16* __restrict__ SWt,   // [8000][1024]
                                                    const float* __restrict__ sb,   // [8000]
                                                    const int* __restrict__ targets,// [16384]
                                                    float* __restrict__ nll) {      // [16384]
  __shared__ bf16  Asl[16 * HH];
  __shared__ float tgt[16];
  __shared__ float pm[4][16];
  __shared__ float ps[4][16];

  const int tid  = threadIdx.x;
  const int wid  = tid >> 5;
  const int lane = tid & 31;
  const int hlf  = lane >> 4;
  const int l16  = lane & 15;
  const int R0   = blockIdx.x * 16;

  // stage A rows into LDS (reused by 4 waves x 500 tiles)
  for (int e = tid * 8; e < 16 * HH; e += 128 * 8) {
    int row = e >> 10;
    int col = e & 1023;
    *(v8bf*)&Asl[e] = *(const v8bf*)(outs + (size_t)(R0 + row) * HH + col);
  }
  __syncthreads();

  int tcol[8];
#pragma unroll
  for (int r = 0; r < 8; ++r) tcol[r] = targets[R0 + r + 8 * hlf];

  float m_run[8], s_run[8];
#pragma unroll
  for (int r = 0; r < 8; ++r) { m_run[r] = -3.0e38f; s_run[r] = 0.0f; }

  for (int p = wid; p < (VV / 64); p += 4) {     // 125 groups of 4 tiles
    const int nb = p * 64 + l16;
    const bf16* bp0 = SWt + (size_t)(nb)      * HH;
    const bf16* bp1 = SWt + (size_t)(nb + 16) * HH;
    const bf16* bp2 = SWt + (size_t)(nb + 32) * HH;
    const bf16* bp3 = SWt + (size_t)(nb + 48) * HH;
    v8f acc0 = {}, acc1 = {}, acc2 = {}, acc3 = {};
#pragma unroll 2
    for (int k0 = 0; k0 < HH; k0 += 32) {
      union { v16bf v; v8bf h[2]; } ua;
      const bf16* ar = &Asl[l16 * HH + k0 + hlf * 16];
      ua.h[0] = *(const v8bf*)ar;
      ua.h[1] = *(const v8bf*)(ar + 8);
      const int ko = k0 + hlf * 16;
      v16bf b0 = *(const v16bf*)(bp0 + ko);
      v16bf b1 = *(const v16bf*)(bp1 + ko);
      v16bf b2 = *(const v16bf*)(bp2 + ko);
      v16bf b3 = *(const v16bf*)(bp3 + ko);
      acc0 = wmma_bf16(ua.v, b0, acc0);
      acc1 = wmma_bf16(ua.v, b1, acc1);
      acc2 = wmma_bf16(ua.v, b2, acc2);
      acc3 = wmma_bf16(ua.v, b3, acc3);
    }
    const float bias0 = sb[nb];
    const float bias1 = sb[nb + 16];
    const float bias2 = sb[nb + 32];
    const float bias3 = sb[nb + 48];
#pragma unroll
    for (int r = 0; r < 8; ++r) {
      float v0 = acc0[r] + bias0;
      float v1 = acc1[r] + bias1;
      float v2 = acc2[r] + bias2;
      float v3 = acc3[r] + bias3;
      const int tc = tcol[r];
      if (nb      == tc) tgt[r + 8 * hlf] = v0;   // unique writer
      if (nb + 16 == tc) tgt[r + 8 * hlf] = v1;
      if (nb + 32 == tc) tgt[r + 8 * hlf] = v2;
      if (nb + 48 == tc) tgt[r + 8 * hlf] = v3;
      // combined 64-col tile: lane-local max of the 4, then 16-lane tree
      float tm = fmaxf(fmaxf(v0, v1), fmaxf(v2, v3));
      for (int off = 1; off < 16; off <<= 1) tm = fmaxf(tm, __shfl_xor(tm, off, 32));
      float ts = expf(v0 - tm) + expf(v1 - tm) + expf(v2 - tm) + expf(v3 - tm);
      for (int off = 1; off < 16; off <<= 1) ts += __shfl_xor(ts, off, 32);
      float nm = fmaxf(m_run[r], tm);
      s_run[r] = s_run[r] * expf(m_run[r] - nm) + ts * expf(tm - nm);
      m_run[r] = nm;
    }
  }

  if (l16 == 0) {
#pragma unroll
    for (int r = 0; r < 8; ++r) {
      pm[wid][r + 8 * hlf] = m_run[r];
      ps[wid][r + 8 * hlf] = s_run[r];
    }
  }
  __syncthreads();

  if (tid < 16) {
    float M = pm[0][tid];
    for (int w = 1; w < 4; ++w) M = fmaxf(M, pm[w][tid]);
    float Ssum = 0.0f;
    for (int w = 0; w < 4; ++w) Ssum += ps[w][tid] * expf(pm[w][tid] - M);
    nll[R0 + tid] = (M + logf(Ssum)) - tgt[tid];
  }
}

// deterministic mean reduction of 16384 per-row NLLs
__global__ __launch_bounds__(256) void k_reduce(const float* __restrict__ nll,
                                                float* __restrict__ out) {
  __shared__ float sm[256];
  float s = 0.0f;
  for (int i = threadIdx.x; i < NROWS; i += 256) s += nll[i];
  sm[threadIdx.x] = s;
  __syncthreads();
  for (int off = 128; off > 0; off >>= 1) {
    if (threadIdx.x < off) sm[threadIdx.x] += sm[threadIdx.x + off];
    __syncthreads();
  }
  if (threadIdx.x == 0) out[0] = sm[0] / (float)NROWS;
}

// =====================================================================
// Host orchestration
// =====================================================================
extern "C" void kernel_launch(void* const* d_in, const int* in_sizes, int n_in,
                              void* d_out, int out_size, void* d_ws, size_t ws_size,
                              hipStream_t stream) {
  const int*   inputs  = (const int*)d_in[0];
  const int*   targets = (const int*)d_in[1];
  const float* emb     = (const float*)d_in[2];
  const float* lstm_w  = (const float*)d_in[3];
  const float* lstm_b  = (const float*)d_in[4];
  const float* sw      = (const float*)d_in[5];
  const float* sbias   = (const float*)d_in[6];
  float*       out     = (float*)d_out;

  char*  ws  = (char*)d_ws;
  size_t off = 0;
  auto alloc = [&](size_t bytes) -> void* {
    void* p = ws + off;
    off = (off + bytes + 255) & ~(size_t)255;
    return p;
  };

  bf16*  x_bf = (bf16*)alloc((size_t)BB * SS * HH * 2);        // 32 MB
  bf16*  wt   = (bf16*)alloc((size_t)LL * N4H * K2H * 2);      // 32 MB
  bf16*  swt  = (bf16*)alloc((size_t)VV * HH * 2);             // 16 MB
  bf16*  outs = (bf16*)alloc((size_t)BB * SS * HH * 2);        // 32 MB
  float* nllb = (float*)alloc((size_t)NROWS * 4);
  // contiguous state block (zeroed each launch for determinism):
  //   c0, c1 (f32) + double-buffered h0[2], h1[2] (bf16)
  unsigned int* state = (unsigned int*)alloc((size_t)2 * BB * HH * 4 + (size_t)4 * BB * HH * 2);
  float* c0 = (float*)state;
  float* c1 = c0 + BB * HH;
  bf16*  h0[2];
  bf16*  h1[2];
  h0[0] = (bf16*)(c1 + BB * HH);
  h0[1] = h0[0] + BB * HH;
  h1[0] = h0[1] + BB * HH;
  h1[1] = h1[0] + BB * HH;
  const int state_u32 = (2 * BB * HH * 4 + 4 * BB * HH * 2) / 4;  // 262144

  // one-time prep
  k_embed     <<<(BB * SS * HH) / 256, 256, 0, stream>>>(inputs, emb, x_bf);
  k_cast_lstmw<<<(LL * K2H * N4H) / 256, 256, 0, stream>>>(lstm_w, wt);
  k_cast_sw   <<<(HH * VV + 255) / 256, 256, 0, stream>>>(sw, swt);
  k_zero_u32  <<<(state_u32 + 255) / 256, 256, 0, stream>>>(state, state_u32);

  // sequential LSTM recurrence: one fused kernel per layer per step
  // (bf16 weights, 32 MB total, stay resident in the 192 MB L2 across steps)
  for (int t = 0; t < SS; ++t) {
    const int rd = t & 1, wr = rd ^ 1;
    // layer 0: input = x_t (row stride S*H), h_prev = h0[rd] -> h0[wr]
    k_lstm_step<<<HH / 16, 256, 0, stream>>>(x_bf + (size_t)t * HH, SS * HH,
                                             h0[rd], wt, lstm_b, c0, h0[wr],
                                             nullptr, 0);
    // layer 1: input = new h0, h_prev = h1[rd] -> h1[wr]; also scatter into outs
    k_lstm_step<<<HH / 16, 256, 0, stream>>>(h0[wr], HH,
                                             h1[rd], wt + (size_t)N4H * K2H,
                                             lstm_b + N4H, c1, h1[wr],
                                             outs + (size_t)t * HH, SS * HH);
  }

  // fused projection + log-softmax + NLL (no 512MB logits buffer)
  k_logits_nll<<<NROWS / 16, 128, 0, stream>>>(outs, swt, sbias, targets, nllb);
  k_reduce    <<<1, 256, 0, stream>>>(nllb, out);
}
